// MPNNLayer_29403346108688
// MI455X (gfx1250) — matrix operand order repaired and checked by reading, not compiled, not run on hardware
//
#include <hip/hip_runtime.h>
#include <hip/hip_bf16.h>

// ---------------------------------------------------------------------------
// MPNN layer for MI455X (gfx1250): bf16 WMMA edge MLP + L2-resident atomics
// v3: fragment halves concatenated via shufflevector (no per-element
//     v_mov_b16 assembly), 32-edge M-tiles, packed bf16 staging stores.
// ---------------------------------------------------------------------------

#define DIN  64
#define DE   64
#define DOUT 64
#define DG   128
#define KIN  192   // 2*DIN + DE

typedef __bf16 bf16;
typedef __attribute__((ext_vector_type(16))) __bf16 v16bf;
typedef __attribute__((ext_vector_type(8)))  __bf16 v8bf;
typedef __attribute__((ext_vector_type(8)))  float  v8f;

__device__ __forceinline__ bf16 f2bf(float f) {
  unsigned u = __builtin_bit_cast(unsigned, f);
  u += 0x7FFFu + ((u >> 16) & 1u);               // round-to-nearest-even
  unsigned short s = (unsigned short)(u >> 16);
  return __builtin_bit_cast(bf16, s);
}

// pack two floats into two RNE bf16 in one dword (lo = a, hi = b)
__device__ __forceinline__ unsigned pack2bf(float a, float b) {
  unsigned ua = __builtin_bit_cast(unsigned, a);
  ua += 0x7FFFu + ((ua >> 16) & 1u);
  unsigned ub = __builtin_bit_cast(unsigned, b);
  ub += 0x7FFFu + ((ub >> 16) & 1u);
  return (ua >> 16) | (ub & 0xFFFF0000u);
}

__device__ __forceinline__ v8f wmma_bf16(v16bf a, v16bf b, v8f c) {
  // v_wmma_f32_16x16x32_bf16: D = A(16x32) x B(32x16) + C(16x16 f32)
  return __builtin_amdgcn_wmma_f32_16x16x32_bf16(false, a, false, b,
                                                 (short)0, c, false, false);
}

// concat two 8-wide halves into one 16-wide fragment (register coalesce,
// no per-element v_mov_b16)
__device__ __forceinline__ v16bf cat8(v8bf lo, v8bf hi) {
  return __builtin_shufflevector(lo, hi, 0, 1, 2, 3, 4, 5, 6, 7,
                                 8, 9, 10, 11, 12, 13, 14, 15);
}

// A-matrix fragment (16x32 bf16).  ISA layout: lanes 0-15 -> rows, elements
// 0..7 hold K = kbase+hi..+7, elements 8..15 hold K = kbase+16+hi..+7,
// hi = 8 for lanes 16-31.  Row data is K-contiguous -> two ds_load_b128.
__device__ __forceinline__ v16bf load_a_frag(const bf16* xrow, int kbase, int hi8) {
  return cat8(*reinterpret_cast<const v8bf*>(xrow + kbase + hi8),
              *reinterpret_cast<const v8bf*>(xrow + kbase + 16 + hi8));
}

// B-matrix fragment (32x16 bf16) from LDS weights stored TRANSPOSED [n][k]:
// lane L holds column n = L&15 with K = kbase + hi..+7 and kbase+16+hi..+7, hi = ((L>>4)<<3).
__device__ __forceinline__ v16bf load_b_frag(const bf16* wt, int n, int ldk,
                                             int kbase, int lane) {
  const bf16* p = wt + n * ldk + kbase + ((lane >> 4) << 3);
  return cat8(*reinterpret_cast<const v8bf*>(p),
              *reinterpret_cast<const v8bf*>(p + 16));
}

// ---------------------------------------------------------------------------
__global__ void zero_kernel(float* __restrict__ p, long n) {
  long i = (long)blockIdx.x * blockDim.x + threadIdx.x;
  if (i < n) p[i] = 0.0f;
}

// ---------------------------------------------------------------------------
// Edge MLP: msg = relu(relu([h_src|h_dst|h_e] @ W1 + b1) @ W2 + b2);
// scatter-sum into agg[dst].  2 waves/block, one 32-edge (2 M-tile) tile per
// wave so every B fragment is reused for two WMMAs.
// LDS: W1t 24KB + W2t 8KB + per-wave X 12KB*2 (Y aliases X) = 56KB.
// ---------------------------------------------------------------------------
__global__ __launch_bounds__(64) void edge_mlp_kernel(
    const float* __restrict__ h_node, const float* __restrict__ h_edge,
    const float* __restrict__ W1, const float* __restrict__ b1,
    const float* __restrict__ W2, const float* __restrict__ b2,
    const int* __restrict__ src, const int* __restrict__ dst,
    float* __restrict__ agg, int E) {
  __shared__ bf16 w1t[DOUT * KIN];    // [n][k]
  __shared__ bf16 w2t[DOUT * DOUT];   // [n][k]
  __shared__ bf16 xs[2][32 * KIN];    // per-wave 32-edge tile; reused for Y

  const int tid  = threadIdx.x;
  const int wave = tid >> 5;
  const int lane = tid & 31;
  const int nlo  = lane & 15;
  const int hi8  = (lane >> 4) << 3;

  // stage transposed weights as bf16, two elements (one dword) per store
  for (int i = tid * 2; i < DOUT * KIN; i += 64 * 2) {
    int n = i / KIN, k = i % KIN;     // k even; k and k+1 in same row
    *reinterpret_cast<unsigned*>(&w1t[i]) =
        pack2bf(W1[k * DOUT + n], W1[(k + 1) * DOUT + n]);
  }
  for (int i = tid * 2; i < DOUT * DOUT; i += 64 * 2) {
    int n = i / DOUT, k = i % DOUT;
    *reinterpret_cast<unsigned*>(&w2t[i]) =
        pack2bf(W2[k * DOUT + n], W2[(k + 1) * DOUT + n]);
  }

  long tileBase = (long)(blockIdx.x * 2 + wave) * 32;
  const bool valid = tileBase < (long)E;
  const long tb = valid ? tileBase : 0;

  // prefetch the dst indices used by the scatter phase at the end
  __builtin_prefetch(dst + tb, 0, 1);

  // gather X tile: 32 edges x 192 features -> bf16 LDS
  // (float4 global loads -> one uint2 (4 x bf16) ds store)
  bf16* x = xs[wave];
  for (int i = lane; i < 32 * 48; i += 32) {
    int er = i / 48;          // edge row in tile
    int c4 = i % 48;          // float4 column
    long e = tb + er;
    const float* p;
    if (c4 < 16)      p = h_node + (long)src[e] * DIN + c4 * 4;
    else if (c4 < 32) p = h_node + (long)dst[e] * DIN + (c4 - 16) * 4;
    else              p = h_edge + e * (long)DE + (c4 - 32) * 4;
    float4 v = *reinterpret_cast<const float4*>(p);
    uint2 pv;
    pv.x = pack2bf(v.x, v.y);
    pv.y = pack2bf(v.z, v.w);
    *reinterpret_cast<uint2*>(x + er * KIN + c4 * 4) = pv;
  }
  __syncthreads();

  // ---- layer 1: Y = relu(X @ W1 + b1), 2 M-tiles x 4 N-tiles ----
  v8f acc[2][4];
#pragma unroll
  for (int nt = 0; nt < 4; ++nt) {
    float bias = b1[nt * 16 + nlo];
#pragma unroll
    for (int r = 0; r < 8; ++r) { acc[0][nt][r] = bias; acc[1][nt][r] = bias; }
  }
#pragma unroll
  for (int kb = 0; kb < 6; ++kb) {
    v16bf a0 = load_a_frag(x + nlo * KIN, kb * 32, hi8);
    v16bf a1 = load_a_frag(x + (16 + nlo) * KIN, kb * 32, hi8);
#pragma unroll
    for (int nt = 0; nt < 4; ++nt) {
      v16bf b = load_b_frag(w1t, nt * 16 + nlo, KIN, kb * 32, lane);
      acc[0][nt] = wmma_bf16(a0, b, acc[0][nt]);
      acc[1][nt] = wmma_bf16(a1, b, acc[1][nt]);
    }
  }

  // ReLU -> bf16 Y tile (32 x 64), aliased into this wave's X region
  bf16* y = xs[wave];
#pragma unroll
  for (int t = 0; t < 2; ++t)
#pragma unroll
    for (int nt = 0; nt < 4; ++nt)
#pragma unroll
      for (int r = 0; r < 8; ++r) {
        float v = acc[t][nt][r];
        v = v > 0.0f ? v : 0.0f;
        y[(t * 16 + r + hi8) * DOUT + nt * 16 + nlo] = f2bf(v);
      }
  __syncthreads();

  // ---- layer 2: M = relu(Y @ W2 + b2) ----
#pragma unroll
  for (int nt = 0; nt < 4; ++nt) {
    float bias = b2[nt * 16 + nlo];
#pragma unroll
    for (int r = 0; r < 8; ++r) { acc[0][nt][r] = bias; acc[1][nt][r] = bias; }
  }
#pragma unroll
  for (int kb = 0; kb < 2; ++kb) {
    v16bf a0 = load_a_frag(y + nlo * DOUT, kb * 32, hi8);
    v16bf a1 = load_a_frag(y + (16 + nlo) * DOUT, kb * 32, hi8);
#pragma unroll
    for (int nt = 0; nt < 4; ++nt) {
      v16bf b = load_b_frag(w2t, nt * 16 + nlo, DOUT, kb * 32, lane);
      acc[0][nt] = wmma_bf16(a0, b, acc[0][nt]);
      acc[1][nt] = wmma_bf16(a1, b, acc[1][nt]);
    }
  }

  // scatter-sum into agg[dst] (agg is 13MB -> L2-resident f32 atomics)
  if (valid) {
#pragma unroll
    for (int t = 0; t < 2; ++t)
#pragma unroll
      for (int r = 0; r < 8; ++r) {
        long e = tb + t * 16 + r + hi8;
        if (e < (long)E) {
          int d = dst[e];
#pragma unroll
          for (int nt = 0; nt < 4; ++nt) {
            float v = acc[t][nt][r];
            v = v > 0.0f ? v : 0.0f;
            atomicAdd(&agg[(long)d * DOUT + nt * 16 + nlo], v);
          }
        }
      }
  }
}

// ---------------------------------------------------------------------------
// Node transform + per-graph pooling: n = agg @ Wn + bn (no ReLU);
// pooled[graph] += n.  8 waves/block, 16 nodes per wave.
// ---------------------------------------------------------------------------
__global__ __launch_bounds__(256) void node_pool_kernel(
    const float* __restrict__ agg, const float* __restrict__ Wn,
    const float* __restrict__ bn, const int* __restrict__ node_graph_id,
    float* __restrict__ pooled, int N) {
  __shared__ bf16 wnt[DOUT * DOUT];   // [n][k]
  __shared__ bf16 xs[8][16 * DOUT];

  const int tid  = threadIdx.x;
  const int wave = tid >> 5;
  const int lane = tid & 31;
  const int nlo  = lane & 15;
  const int hi8  = (lane >> 4) << 3;

  for (int i = tid * 2; i < DOUT * DOUT; i += 256 * 2) {
    int n = i / DOUT, k = i % DOUT;
    *reinterpret_cast<unsigned*>(&wnt[i]) =
        pack2bf(Wn[k * DOUT + n], Wn[(k + 1) * DOUT + n]);
  }

  long tileBase = (long)(blockIdx.x * 8 + wave) * 16;
  const bool valid = tileBase < (long)N;
  const long tb = valid ? tileBase : 0;

  __builtin_prefetch(node_graph_id + tb, 0, 1);

  bf16* x = xs[wave];
  for (int i = lane; i < 16 * 16; i += 32) {   // 16 nodes x 16 float4
    int nr = i / 16, c4 = i % 16;
    float4 v = *reinterpret_cast<const float4*>(agg + (tb + nr) * (long)DOUT + c4 * 4);
    uint2 pv;
    pv.x = pack2bf(v.x, v.y);
    pv.y = pack2bf(v.z, v.w);
    *reinterpret_cast<uint2*>(x + nr * DOUT + c4 * 4) = pv;
  }
  __syncthreads();

  v8f acc[4];
#pragma unroll
  for (int nt = 0; nt < 4; ++nt) {
    float bias = bn[nt * 16 + nlo];
#pragma unroll
    for (int r = 0; r < 8; ++r) acc[nt][r] = bias;
  }
#pragma unroll
  for (int kb = 0; kb < 2; ++kb) {
    v16bf a = load_a_frag(x + nlo * DOUT, kb * 32, hi8);
#pragma unroll
    for (int nt = 0; nt < 4; ++nt) {
      v16bf b = load_b_frag(wnt, nt * 16 + nlo, DOUT, kb * 32, lane);
      acc[nt] = wmma_bf16(a, b, acc[nt]);
    }
  }

  if (valid) {
#pragma unroll
    for (int r = 0; r < 8; ++r) {
      long node = tb + r + hi8;
      if (node < (long)N) {
        int g = node_graph_id[node];
#pragma unroll
        for (int nt = 0; nt < 4; ++nt)
          atomicAdd(&pooled[(long)g * DOUT + nt * 16 + nlo], acc[nt][r]);
      }
    }
  }
}

// ---------------------------------------------------------------------------
// Global MLP: out = relu([u | pooled] @ Wg + bg).  One block, 2 waves,
// each wave handles 4 of the 8 sixteen-graph M tiles.  Wg^T in LDS (48KB).
// ---------------------------------------------------------------------------
__global__ __launch_bounds__(64) void global_mlp_kernel(
    const float* __restrict__ u, const float* __restrict__ pooled,
    const float* __restrict__ Wg, const float* __restrict__ bg,
    float* __restrict__ out) {
  __shared__ bf16 wgt[DG * KIN];      // [n][k] 128x192
  __shared__ bf16 xs[2][16 * KIN];

  const int tid  = threadIdx.x;
  const int wave = tid >> 5;
  const int lane = tid & 31;
  const int nlo  = lane & 15;
  const int hi8  = (lane >> 4) << 3;

  for (int i = tid * 2; i < DG * KIN; i += 64 * 2) {
    int n = i / KIN, k = i % KIN;
    *reinterpret_cast<unsigned*>(&wgt[i]) =
        pack2bf(Wg[k * DG + n], Wg[(k + 1) * DG + n]);
  }
  __syncthreads();

  for (int mt = wave; mt < 8; mt += 2) {
    bf16* x = xs[wave];
    for (int i = lane * 2; i < 16 * KIN; i += 32 * 2) {
      int gr = i / KIN, c = i % KIN;  // c even; pair never straddles u/pooled
      int g  = mt * 16 + gr;
      float v0, v1;
      if (c < DG) { v0 = u[(long)g * DG + c];  v1 = u[(long)g * DG + c + 1]; }
      else { v0 = pooled[(long)g * DOUT + (c - DG)];
             v1 = pooled[(long)g * DOUT + (c - DG) + 1]; }
      *reinterpret_cast<unsigned*>(x + i) = pack2bf(v0, v1);
    }
    // per-wave LDS region; same-wave DS ops are in-order

    v8f acc[8];
#pragma unroll
    for (int nt = 0; nt < 8; ++nt) {
      float bias = bg[nt * 16 + nlo];
#pragma unroll
      for (int r = 0; r < 8; ++r) acc[nt][r] = bias;
    }
#pragma unroll
    for (int kb = 0; kb < 6; ++kb) {
      v16bf a = load_a_frag(x + nlo * KIN, kb * 32, hi8);
#pragma unroll
      for (int nt = 0; nt < 8; ++nt) {
        v16bf b = load_b_frag(wgt, nt * 16 + nlo, KIN, kb * 32, lane);
        acc[nt] = wmma_bf16(a, b, acc[nt]);
      }
    }
#pragma unroll
    for (int nt = 0; nt < 8; ++nt)
#pragma unroll
      for (int r = 0; r < 8; ++r) {
        float v = acc[nt][r];
        v = v > 0.0f ? v : 0.0f;
        out[(long)(mt * 16 + r + hi8) * DG + nt * 16 + nlo] = v;
      }
  }
}

// ---------------------------------------------------------------------------
extern "C" void kernel_launch(void* const* d_in, const int* in_sizes, int n_in,
                              void* d_out, int out_size, void* d_ws, size_t ws_size,
                              hipStream_t stream) {
  const float* h_node = (const float*)d_in[0];
  const float* h_edge = (const float*)d_in[1];
  const float* u      = (const float*)d_in[2];
  const float* W1     = (const float*)d_in[3];
  const float* b1     = (const float*)d_in[4];
  const float* W2     = (const float*)d_in[5];
  const float* b2     = (const float*)d_in[6];
  const float* Wn     = (const float*)d_in[7];
  const float* bn     = (const float*)d_in[8];
  const float* Wg     = (const float*)d_in[9];
  const float* bg     = (const float*)d_in[10];
  const int*   src    = (const int*)d_in[11];
  const int*   dst    = (const int*)d_in[12];
  const int*   ngi    = (const int*)d_in[13];

  const int N = in_sizes[0] / DIN;   // 51200
  const int E = in_sizes[1] / DE;    // 800000
  const int G = in_sizes[2] / DG;    // 128

  float* agg    = (float*)d_ws;                       // [N, DOUT]
  float* pooled = agg + (size_t)N * DOUT;             // [G, DOUT]

  long zn = (long)N * DOUT + (long)G * DOUT;
  zero_kernel<<<(int)((zn + 255) / 256), 256, 0, stream>>>(agg, zn);

  int edgeBlocks = (E + 63) / 64;                     // 2 waves x 32 edges
  edge_mlp_kernel<<<edgeBlocks, 64, 0, stream>>>(h_node, h_edge, W1, b1, W2,
                                                 b2, src, dst, agg, E);

  int nodeBlocks = (N + 127) / 128;                   // 8 waves x 16 nodes
  node_pool_kernel<<<nodeBlocks, 256, 0, stream>>>(agg, Wn, bn, ngi, pooled, N);

  global_mlp_kernel<<<1, 64, 0, stream>>>(u, pooled, Wg, bg, (float*)d_out);
  (void)G; (void)n_in; (void)out_size; (void)ws_size;
}
